// SubjectGenerator_72395968742088
// MI455X (gfx1250) — matrix operand order repaired and checked by reading, not compiled, not run on hardware
//
#include <hip/hip_runtime.h>
#include <hip/hip_bf16.h>

typedef __bf16 bf16_t;
typedef __attribute__((ext_vector_type(16))) __bf16 v16bf;
typedef __attribute__((ext_vector_type(8)))  float  v8f;

#define T_STEPS 2048
#define BATCH   64
#define IN_DIM  27
#define H1      400
#define H2      200

// K layout for gates GEMM: [x padded 27->32][h padded 400->416]
#define KX        32
#define KH        416
#define KTILES_G  14   // 1 x-tile + 13 h-tiles of K=32
#define KTILES_F  13   // fc: K=400 padded to 416
#define MT_G      100  // 1600 gate rows / 16  (= 4 gates x 25 h-tiles)
#define MT_F      13   // 200 fc rows padded to 208 / 16
#define FRAG      512  // elements in one 16x32 A fragment

#define NG (MT_G * KTILES_G * FRAG)
#define NF (MT_F * KTILES_F * FRAG)

#define NTHREADS 512   // 16 waves: 4 waves/SIMD -> ~2x VGPR budget, no spills

// A-matrix 16x32 layout (contiguous, matches B loads): element e -> K = e;
// lane>=16 adds +16.
__host__ __device__ __forceinline__ int kmap(int e, int hi) {
  return e + (hi ? 16 : 0);
}

// One-shot weight repack: fp32 -> bf16, fragment-ordered for WMMA A operand.
__global__ void pack_weights(const float* __restrict__ W_ih,
                             const float* __restrict__ W_hh,
                             const float* __restrict__ W_fc,
                             bf16_t* __restrict__ wsg,
                             bf16_t* __restrict__ wsf) {
  const int total = NG + NF;
  for (int i = blockIdx.x * blockDim.x + threadIdx.x; i < total;
       i += gridDim.x * blockDim.x) {
    if (i < NG) {
      int frag = i >> 9, r = i & 511;
      int lane = r >> 4, e = r & 15;
      int mt = frag / KTILES_G, kt = frag % KTILES_G;
      int row = mt * 16 + (lane & 15);        // gate row 0..1599
      int hi  = lane >> 4;
      int kk  = kt * 32 + kmap(e, hi);        // combined K 0..447
      float v = 0.f;
      if (kk < KX) {
        if (kk < IN_DIM) v = W_ih[row * IN_DIM + kk];
      } else {
        int c = kk - KX;
        if (c < H1) v = W_hh[row * H1 + c];
      }
      wsg[i] = (bf16_t)v;
    } else {
      int j = i - NG;
      int frag = j >> 9, r = j & 511;
      int lane = r >> 4, e = r & 15;
      int mt = frag / KTILES_F, kt = frag % KTILES_F;
      int row = mt * 16 + (lane & 15);
      int hi  = lane >> 4;
      int col = kt * 32 + kmap(e, hi);
      float v = (row < H2 && col < H1) ? W_fc[row * H1 + col] : 0.f;
      wsf[j] = (bf16_t)v;
    }
  }
}

__global__ __launch_bounds__(NTHREADS, 1)
void lstm_persistent(const float* __restrict__ word,
                     const bf16_t* __restrict__ wsg,
                     const bf16_t* __restrict__ wsf,
                     const float* __restrict__ b_ih,
                     const float* __restrict__ b_hh,
                     const float* __restrict__ b_fc,
                     const float* __restrict__ W_out,
                     const float* __restrict__ b_out,
                     float* __restrict__ out) {
  extern __shared__ char smem[];
  bf16_t* xbuf   = (bf16_t*)smem;                      // [2][64][32]  bf16
  bf16_t* hbuf   = xbuf + 2 * BATCH * KX;              // [2][64][416] bf16
  float*  fc_lds = (float*)(hbuf + 2 * BATCH * KH);    // [208][64] f32
  float*  cbuf   = fc_lds + 208 * BATCH;               // [400][64] f32 cell state
  float*  bias_g  = cbuf + H1 * BATCH;                 // [1600]
  float*  bias_fc = bias_g + 4 * H1;                   // [208]
  float*  wout    = bias_fc + 208;                     // [200]

  const int tid    = threadIdx.x;
  const int wave   = tid >> 5;
  const int lane   = tid & 31;
  const int lanelo = lane & 15;
  const int hi     = lane >> 4;

  // ---- one-time init ----
  for (int r = tid; r < 4 * H1; r += NTHREADS) bias_g[r] = b_ih[r] + b_hh[r];
  for (int r = tid; r < 208; r += NTHREADS)    bias_fc[r] = (r < H2) ? b_fc[r] : 0.f;
  for (int r = tid; r < H2; r += NTHREADS)     wout[r] = W_out[r];
  for (int r = tid; r < 2 * BATCH * KH; r += NTHREADS) hbuf[r] = (bf16_t)0.f;
  for (int r = tid; r < H1 * BATCH; r += NTHREADS)     cbuf[r] = 0.f;
  // stage x(0) into xbuf[0]
  for (int i = tid; i < BATCH * KX; i += NTHREADS) {
    int b = i >> 5, c = i & 31;
    xbuf[i] = (bf16_t)((c < IN_DIM) ? word[b * IN_DIM + c] : 0.f);
  }
  __syncthreads();

  const float bo = b_out[0];
  const size_t gstride = (size_t)25 * KTILES_G * FRAG;  // per-gate frag block

  for (int t = 0; t < T_STEPS; ++t) {
    const int p = t & 1;
    const bf16_t* xb = xbuf + p * BATCH * KX;
    const bf16_t* hp = hbuf + p * BATCH * KH;        // h from previous step
    bf16_t*       hn = hbuf + (p ^ 1) * BATCH * KH;  // h produced this step

    // ================= Phase A: gates (+ overlapped head / prefetch) ========
    // waves 0..8 own two h-tiles (ht = w, w+16); waves 9..15 own one.
    const int nunits = (wave < 9) ? 2 : 1;
    for (int uu = 0; uu < nunits; ++uu) {
      const int ht = wave + uu * 16;           // 0..24
      v8f acc[4][4];                           // [gate][bt] = 128 VGPRs
      v8f zero = {};
#pragma unroll
      for (int g4 = 0; g4 < 4; ++g4)
#pragma unroll
        for (int bt = 0; bt < 4; ++bt) acc[g4][bt] = zero;

      for (int kt = 0; kt < KTILES_G; ++kt) {
        v16bf bfr[4];
#pragma unroll
        for (int bt = 0; bt < 4; ++bt) {
          int brow = bt * 16 + lanelo;
          bfr[bt] = (kt == 0)
            ? *(const v16bf*)(xb + brow * KX + hi * 16)
            : *(const v16bf*)(hp + brow * KH + (kt - 1) * 32 + hi * 16);
        }
        const bf16_t* base = wsg + ((size_t)ht * KTILES_G + kt) * FRAG + lane * 16;
#pragma unroll
        for (int g4 = 0; g4 < 4; ++g4) {
          v16bf wa = *(const v16bf*)(base + g4 * gstride);
          acc[g4][0] = __builtin_amdgcn_wmma_f32_16x16x32_bf16(false, wa, false, bfr[0], (short)0, acc[g4][0], false, false);
          acc[g4][1] = __builtin_amdgcn_wmma_f32_16x16x32_bf16(false, wa, false, bfr[1], (short)0, acc[g4][1], false, false);
          acc[g4][2] = __builtin_amdgcn_wmma_f32_16x16x32_bf16(false, wa, false, bfr[2], (short)0, acc[g4][2], false, false);
          acc[g4][3] = __builtin_amdgcn_wmma_f32_16x16x32_bf16(false, wa, false, bfr[3], (short)0, acc[g4][3], false, false);
        }
      }
      // LSTM elementwise; cell state lives in LDS (f32, full precision).
      const int mbase = ht * 16 + hi * 8;
#pragma unroll
      for (int bt = 0; bt < 4; ++bt) {
        const int brow = bt * 16 + lanelo;
#pragma unroll
        for (int g = 0; g < 8; ++g) {
          int m = mbase + g;                       // hidden unit 0..399
          float xi = acc[0][bt][g] + bias_g[m];
          float xf = acc[1][bt][g] + bias_g[H1 + m];
          float xg = acc[2][bt][g] + bias_g[2 * H1 + m];
          float xo = acc[3][bt][g] + bias_g[3 * H1 + m];
          float iv = 1.f / (1.f + __expf(-xi));
          float fv = 1.f / (1.f + __expf(-xf));
          float gv = tanhf(xg);
          float ov = 1.f / (1.f + __expf(-xo));
          float cn = fv * cbuf[m * BATCH + brow] + iv * gv;
          float hv = ov * tanhf(cn);
          cbuf[m * BATCH + brow] = fmaxf(cn, 0.f);       // relu1 on cell
          hn[brow * KH + m] = (bf16_t)fmaxf(hv, 0.f);    // relu1 on hidden
        }
      }
    }
    if (wave == 13) {
      // prefetch next step's input rows (emits global_prefetch_b8)
      if (t + 1 < T_STEPS) {
        const float* wp = word + (size_t)(t + 1) * BATCH * IN_DIM;
        __builtin_prefetch(wp + lane * 32, 0, 3);
        if (lane < 22) __builtin_prefetch(wp + (32 + lane) * 32, 0, 3);
      }
    } else if (wave >= 14) {
      // output head for step t-1, overlapped with gates(t): 64 lanes
      if (t > 0) {
        int b = (wave - 14) * 32 + lane;      // 0..63
        float y = bo;
        for (int h = 0; h < H2; ++h) y += fc_lds[h * BATCH + b] * wout[h];
        out[(t - 1) * BATCH + b] = 1.f / (1.f + __expf(-y));
      }
    }
    __syncthreads();

    // ================= Phase B: fc GEMM + x staging =================
    if (wave < 13) {
      const int mt = wave;
      v8f fa[4];
      v8f zero = {};
#pragma unroll
      for (int bt = 0; bt < 4; ++bt) fa[bt] = zero;
      for (int kt = 0; kt < KTILES_F; ++kt) {
        v16bf wa = *(const v16bf*)(wsf + ((size_t)mt * KTILES_F + kt) * FRAG + lane * 16);
#pragma unroll
        for (int bt = 0; bt < 4; ++bt) {
          v16bf bfr = *(const v16bf*)(hn + (bt * 16 + lanelo) * KH + kt * 32 + hi * 16);
          fa[bt] = __builtin_amdgcn_wmma_f32_16x16x32_bf16(false, wa, false, bfr, (short)0, fa[bt], false, false);
        }
      }
      const int m0 = mt * 16 + hi * 8;
#pragma unroll
      for (int bt = 0; bt < 4; ++bt)
#pragma unroll
        for (int g = 0; g < 8; ++g)
          fc_lds[(m0 + g) * BATCH + bt * 16 + lanelo] =
              fmaxf(fa[bt][g] + bias_fc[m0 + g], 0.f);
    } else {
      // waves 13..15: stage x(t+1) into the other x buffer
      if (t + 1 < T_STEPS) {
        bf16_t* xn = xbuf + (p ^ 1) * BATCH * KX;
        const float* wsrc = word + (size_t)(t + 1) * BATCH * IN_DIM;
        for (int i = tid - 13 * 32; i < BATCH * KX; i += 3 * 32) {
          int b = i >> 5, c = i & 31;
          xn[i] = (bf16_t)((c < IN_DIM) ? wsrc[b * IN_DIM + c] : 0.f);
        }
      }
    }
    __syncthreads();
  }

  // final output head for t = T-1
  if (tid < BATCH) {
    float y = bo;
    for (int h = 0; h < H2; ++h) y += fc_lds[h * BATCH + tid] * wout[h];
    out[(T_STEPS - 1) * BATCH + tid] = 1.f / (1.f + __expf(-y));
  }
}

extern "C" void kernel_launch(void* const* d_in, const int* in_sizes, int n_in,
                              void* d_out, int out_size, void* d_ws, size_t ws_size,
                              hipStream_t stream) {
  const float* word  = (const float*)d_in[0];
  const float* W_ih  = (const float*)d_in[1];
  const float* W_hh  = (const float*)d_in[2];
  const float* b_ih  = (const float*)d_in[3];
  const float* b_hh  = (const float*)d_in[4];
  const float* W_fc  = (const float*)d_in[5];
  const float* b_fc  = (const float*)d_in[6];
  const float* W_out = (const float*)d_in[7];
  const float* b_out = (const float*)d_in[8];

  bf16_t* wsg = (bf16_t*)d_ws;      // ~1.43 MB gate-weight fragments
  bf16_t* wsf = wsg + NG;           // ~0.17 MB fc-weight fragments

  pack_weights<<<512, 256, 0, stream>>>(W_ih, W_hh, W_fc, wsg, wsf);

  size_t smem = (size_t)(2 * BATCH * KX + 2 * BATCH * KH) * sizeof(bf16_t)
              + (size_t)(208 * BATCH + H1 * BATCH + 4 * H1 + 208 + H2) * sizeof(float);
  lstm_persistent<<<1, NTHREADS, smem, stream>>>(word, wsg, wsf, b_ih, b_hh,
                                                 b_fc, W_out, b_out, (float*)d_out);
}
